// WindowAttention_16286515986820
// MI455X (gfx1250) — compile-verified, matrix operand order checked by reading, not run on hardware
//
#include <hip/hip_runtime.h>
#include <hip/hip_bf16.h>

#define WIN   7
#define NH    4
#define DIM   128
#define HD    32
#define NTOK  49
#define NPAD  64
#define NWIN  1024

typedef __attribute__((ext_vector_type(16))) _Float16 v16h;
typedef __attribute__((ext_vector_type(8)))  _Float16 v8h;
typedef __attribute__((ext_vector_type(8)))  float    v8f;
typedef __attribute__((ext_vector_type(4)))  float    v4f;

// ---------------- LDS layout (bytes) ----------------
// region0: union { xs[64][136] f16 (17408B) ; attn[4][64][68] f32 (69632B) }
// k   : 64*136*2   = 17408   (k[token][chan], row-major)
// q/o : 64*136*2   = 17408   (q[token][chan]; later reused as AV output o[token][chan])
// vT  : 128*72*2   = 18432   (V transposed: vT[chan][token])
// prob: 4*64*72*2  = 36864   (probs[h][token][key], f16)
#define XS_S   136
#define QK_S   136
#define VT_S   72
#define P_S    72
#define AT_S   68
#define OFF_XS    0
#define OFF_ATTN  0
#define OFF_K     69632
#define OFF_Q     (69632 + 17408)
#define OFF_VT    (69632 + 2*17408)
#define OFF_PROBS (69632 + 2*17408 + 18432)
#define SMEM_BYTES (OFF_PROBS + 36864)   // 159744 B

// workspace layout (bytes)
#define WS_WF    0                        // 4*128*128 f16  (Wq pre-scaled by 1/sqrt(32))
#define WS_RB    (4 * DIM * DIM * 2)      // 4*49*49 f32
#define WS_BQS   (WS_RB + NH * NTOK * NTOK * 4)  // 128 f32 (bq pre-scaled)

__device__ __forceinline__ v16h ld16h(const _Float16* p) {
    v16h r;
    *((v8h*)&r)     = *(const v8h*)(p);
    *((v8h*)&r + 1) = *(const v8h*)(p + 8);
    return r;
}

__device__ __forceinline__ v8f wmma16(v16h a, v16h b, v8f c) {
    // v_wmma_f32_16x16x32_f16 : D = A(16x32 f16) * B(32x16 f16) + C(16x16 f32)
    return __builtin_amdgcn_wmma_f32_16x16x32_f16(
        /*neg_a=*/false, a, /*neg_b=*/false, b,
        /*c_mod=*/(short)0, c, /*reuse_a=*/false, /*reuse_b=*/false);
}

// ---------------------------------------------------------------------------
// Prep: f32 weights -> f16 workspace (Wq folded with attention scale);
//       rel_bias -> dense [4][49][49]; bq -> pre-scaled f32 copy.
// ---------------------------------------------------------------------------
__global__ void swin_prep_kernel(const float* __restrict__ Wq, const float* __restrict__ Wk,
                                 const float* __restrict__ Wv, const float* __restrict__ Wp,
                                 const float* __restrict__ rel_bias, const float* __restrict__ bq,
                                 _Float16* __restrict__ wf, float* __restrict__ rbias,
                                 float* __restrict__ bqs) {
    const float scale = 0.17677669529663687f;  // 1/sqrt(32)
    int t = blockIdx.x * 256 + threadIdx.x;
    if (t < 4 * DIM * DIM) {
        const float* src = (t < 16384) ? Wq : (t < 32768) ? Wk : (t < 49152) ? Wv : Wp;
        float s = (t < 16384) ? scale : 1.0f;
        wf[t] = (_Float16)(src[t & 16383] * s);
    }
    if (t < NH * NTOK * NTOK) {
        int h  = t / (NTOK * NTOK);
        int nm = t % (NTOK * NTOK);
        int n = nm / NTOK, m = nm % NTOK;
        int in_ = n / WIN, jn = n % WIN;
        int im  = m / WIN, jm = m % WIN;
        int idx = (in_ - im + WIN - 1) * (2 * WIN - 1) + (jn - jm + WIN - 1);
        rbias[t] = rel_bias[idx * NH + h];
    }
    if (t < DIM) bqs[t] = bq[t] * scale;
}

// ---------------------------------------------------------------------------
// Main: one workgroup (8 waves, wave32) per window. Whole block in LDS.
// Every WMMA stage is oriented so the C-layout (N fixed per lane, M
// consecutive per VGPR) maps to ONE packed 16B store per tile.
// ---------------------------------------------------------------------------
__global__ void __launch_bounds__(256)
swin_attn_kernel(const float* __restrict__ x, const float* __restrict__ mask,
                 const _Float16* __restrict__ wf, const float* __restrict__ rbias,
                 const float* __restrict__ bqs, const float* __restrict__ bk,
                 const float* __restrict__ bv, const float* __restrict__ bp,
                 float* __restrict__ out) {
    extern __shared__ char smem[];
    _Float16* xs    = (_Float16*)(smem + OFF_XS);
    float*    attn  = (float*)   (smem + OFF_ATTN);
    _Float16* kl    = (_Float16*)(smem + OFF_K);
    _Float16* ql    = (_Float16*)(smem + OFF_Q);     // also AV-output staging
    _Float16* vT    = (_Float16*)(smem + OFF_VT);
    _Float16* probs = (_Float16*)(smem + OFF_PROBS);

    const int b    = blockIdx.x;
    const int t    = threadIdx.x;
    const int wave = t >> 5;
    const int lane = t & 31;
    const int lr   = lane & 15;   // N-lane within tile
    const int hi   = lane >> 4;   // half-select (M offset 0/8, K offset 0/16)

    // ---- Phase 0: stream x window in (nontemporal), f32->f16, pad rows ----
    {
        int r  = t >> 2;            // 0..63
        int c0 = (t & 3) * 32;      // 0,32,64,96
        _Float16* dst = xs + r * XS_S + c0;
        if (r < NTOK) {
            const float* src = x + ((size_t)b * NTOK + r) * DIM + c0;
#pragma unroll
            for (int i = 0; i < 4; ++i) {
                v4f f0 = __builtin_nontemporal_load((const v4f*)(src + i * 8));
                v4f f1 = __builtin_nontemporal_load((const v4f*)(src + i * 8 + 4));
                v8h h8;
#pragma unroll
                for (int e = 0; e < 4; ++e) { h8[e] = (_Float16)f0[e]; h8[e + 4] = (_Float16)f1[e]; }
                *(v8h*)(dst + i * 8) = h8;
            }
        } else {
            v8h z = {};
#pragma unroll
            for (int i = 0; i < 4; ++i) *(v8h*)(dst + i * 8) = z;
        }
    }
    __syncthreads();

    // ---- Phase 1a: Q,K as Wᵀ·x -> D(M=chan o, N=token n) ----
    // Scale already folded into Wq/bqs, so Q and K share one branchless
    // epilogue; W/bias/dst selected by uniform pointer cndmask.
#pragma unroll 1
    for (int job = wave; job < 64; job += 8) {
        int m_idx = job >> 5;                 // 0=Q, 1=K
        int tile  = job & 31;
        int ot = tile >> 2, nt = tile & 3;    // 8 chan-tiles x 4 token-tiles
        const _Float16* W   = wf + m_idx * (DIM * DIM);
        const float*    bi  = m_idx ? bk : bqs;
        _Float16*       dst = m_idx ? kl : ql;
        v8f acc = {};
#pragma unroll
        for (int kc = 0; kc < 4; ++kc) {
            v16h a  = ld16h(W  + (ot * 16 + lr) * DIM  + kc * 32 + hi * 16);
            v16h bm = ld16h(xs + (nt * 16 + lr) * XS_S + kc * 32 + hi * 16);
            acc = wmma16(a, bm, acc);
        }
        v4f b0 = *(const v4f*)(bi + ot * 16 + hi * 8);
        v4f b1 = *(const v4f*)(bi + ot * 16 + hi * 8 + 4);
        v8h r;
#pragma unroll
        for (int e = 0; e < 4; ++e) {
            r[e]     = (_Float16)(acc[e]     + b0[e]);
            r[e + 4] = (_Float16)(acc[e + 4] + b1[e]);
        }
        *(v8h*)(dst + (nt * 16 + lr) * QK_S + ot * 16 + hi * 8) = r;
    }

    // ---- Phase 1b: V as x·Wvᵀ -> D(M=token n, N=chan o); store into vT ----
    // Per lane: chan o fixed, 8 consecutive tokens -> packed row of vT[o][*].
#pragma unroll 1
    for (int job = wave; job < 32; job += 8) {
        int mt = job >> 3, ct = job & 7;      // 4 token-tiles x 8 chan-tiles
        const _Float16* W = wf + 2 * (DIM * DIM);
        v8f acc = {};
#pragma unroll
        for (int kc = 0; kc < 4; ++kc) {
            v16h a  = ld16h(xs + (mt * 16 + lr) * XS_S + kc * 32 + hi * 16);
            v16h bm = ld16h(W  + (ct * 16 + lr) * DIM  + kc * 32 + hi * 16);
            acc = wmma16(a, bm, acc);
        }
        int o = ct * 16 + lr;
        float bb = bv[o];
        v8h r;
#pragma unroll
        for (int e = 0; e < 8; ++e) r[e] = (_Float16)(acc[e] + bb);
        *(v8h*)(vT + o * VT_S + mt * 16 + hi * 8) = r;
    }
    __syncthreads();

    // ---- Phase 2: scores as K·Qᵀ -> D(M=key m, N=query n); K=32, 1 WMMA ----
    // Per lane: query row n fixed, 8 consecutive keys -> two packed v4f stores.
#pragma unroll 1
    for (int job = wave; job < 64; job += 8) {
        int h = job >> 4, ij = job & 15, i = ij >> 2, j = ij & 3;  // i: key tile, j: query tile
        v16h a  = ld16h(kl + (i * 16 + lr) * QK_S + h * HD + hi * 16);
        v16h bm = ld16h(ql + (j * 16 + lr) * QK_S + h * HD + hi * 16);
        v8f acc = {};
        acc = wmma16(a, bm, acc);
        float* dst = attn + (h * NPAD + j * 16 + lr) * AT_S + i * 16 + hi * 8;
        v4f s0, s1;
#pragma unroll
        for (int e = 0; e < 4; ++e) { s0[e] = acc[e]; s1[e] = acc[e + 4]; }
        *(v4f*)dst       = s0;
        *(v4f*)(dst + 4) = s1;
    }
    __syncthreads();

    // ---- Phase 3: softmax rows (fused rel-bias + window mask), via LDS ----
    {
        int h = t >> 6, n = t & 63;        // 256 threads = 4 heads x 64 rows
        _Float16* prow = probs + (h * NPAD + n) * P_S;
        if (n < NTOK) {
            float* arow = attn + (h * NPAD + n) * AT_S;
            const float* rrow = rbias + (h * NTOK + n) * NTOK;
            const float* mrow = mask + ((size_t)(b & (NWIN - 1)) * NTOK + n) * NTOK;
            float mx = -1e30f;
            for (int m = 0; m < NTOK; ++m) {
                float v = arow[m] + rrow[m] + mrow[m];
                arow[m] = v;
                mx = fmaxf(mx, v);
            }
            float s = 0.f;
            for (int m = 0; m < NTOK; ++m) {
                float e = __expf(arow[m] - mx);
                arow[m] = e;
                s += e;
            }
            float inv = 1.f / s;
            for (int m = 0; m < NTOK; ++m) prow[m] = (_Float16)(arow[m] * inv);
            for (int m = NTOK; m < NPAD; ++m) prow[m] = (_Float16)0.f;
        } else {
            v8h z = {};
#pragma unroll
            for (int m = 0; m < NPAD; m += 8) *(v8h*)(prow + m) = z;
        }
    }
    __syncthreads();

    // ---- Phase 4: AV as Vᵀ·Pᵀ -> D(M=chan d, N=token n); K=64, 2 WMMAs ----
    // A lane: vT row d (contiguous keys); B lane: probs row n (contiguous keys).
    // Per lane: token n fixed, 8 consecutive chans -> packed store into o[n][d].
#pragma unroll 1
    for (int job = wave; job < 32; job += 8) {
        int h = job >> 3, ic = job & 7, c = ic >> 2, i = ic & 3;   // c: chan tile/2, i: token tile
        v8f acc = {};
#pragma unroll
        for (int kc = 0; kc < 2; ++kc) {
            v16h a  = ld16h(vT    + (h * HD   + c * 16 + lr) * VT_S + kc * 32 + hi * 16);
            v16h bm = ld16h(probs + (h * NPAD + i * 16 + lr) * P_S  + kc * 32 + hi * 16);
            acc = wmma16(a, bm, acc);
        }
        v8h r;
#pragma unroll
        for (int e = 0; e < 8; ++e) r[e] = (_Float16)acc[e];
        *(v8h*)(ql + (i * 16 + lr) * QK_S + h * HD + c * 16 + hi * 8) = r;  // o[n][d]
    }
    __syncthreads();

    // ---- Phase 5: proj as Wp·oᵀ -> D(M=chan o, N=token n); packed HBM out ----
#pragma unroll 1
    for (int job = wave; job < 32; job += 8) {
        int ct = job >> 2, nt = job & 3;      // 8 chan-tiles x 4 token-tiles
        const _Float16* W = wf + 3 * (DIM * DIM);
        v8f acc = {};
#pragma unroll
        for (int kc = 0; kc < 4; ++kc) {
            v16h a  = ld16h(W  + (ct * 16 + lr) * DIM  + kc * 32 + hi * 16);
            v16h bm = ld16h(ql + (nt * 16 + lr) * QK_S + kc * 32 + hi * 16);
            acc = wmma16(a, bm, acc);
        }
        int n = nt * 16 + lr;                 // token: fixed per lane
        if (n < NTOK) {
            v4f b0 = *(const v4f*)(bp + ct * 16 + hi * 8);
            v4f b1 = *(const v4f*)(bp + ct * 16 + hi * 8 + 4);
            v4f s0, s1;
#pragma unroll
            for (int e = 0; e < 4; ++e) { s0[e] = acc[e] + b0[e]; s1[e] = acc[e + 4] + b1[e]; }
            float* op = out + ((size_t)b * NTOK + n) * DIM + ct * 16 + hi * 8;
            __builtin_nontemporal_store(s0, (v4f*)op);
            __builtin_nontemporal_store(s1, (v4f*)(op + 4));
        }
    }
}

// ---------------------------------------------------------------------------
extern "C" void kernel_launch(void* const* d_in, const int* in_sizes, int n_in,
                              void* d_out, int out_size, void* d_ws, size_t ws_size,
                              hipStream_t stream) {
    (void)in_sizes; (void)n_in; (void)out_size; (void)ws_size;
    const float* x    = (const float*)d_in[0];
    const float* mask = (const float*)d_in[1];
    const float* Wq   = (const float*)d_in[2];
    const float* bq   = (const float*)d_in[3];
    const float* Wk   = (const float*)d_in[4];
    const float* bk   = (const float*)d_in[5];
    const float* Wv   = (const float*)d_in[6];
    const float* bv   = (const float*)d_in[7];
    const float* Wp   = (const float*)d_in[8];
    const float* bp   = (const float*)d_in[9];
    const float* relb = (const float*)d_in[10];

    _Float16* wf  = (_Float16*)((char*)d_ws + WS_WF);
    float*    rb  = (float*)((char*)d_ws + WS_RB);
    float*    bqs = (float*)((char*)d_ws + WS_BQS);

    swin_prep_kernel<<<256, 256, 0, stream>>>(Wq, Wk, Wv, Wp, relb, bq, wf, rb, bqs);

    const int n_windows = 16384;
    swin_attn_kernel<<<n_windows, 256, SMEM_BYTES, stream>>>(
        x, mask, wf, rb, bqs, bk, bv, bp, (float*)d_out);
}
